// Step3SchNet_35905926595305
// MI455X (gfx1250) — matrix operand order repaired and checked by reading, not compiled, not run on hardware
//
#include <hip/hip_runtime.h>
#include <hip/hip_bf16.h>
#include <math.h>

#define N_ATOMS 50000
#define N_PAD   50048              // padded to a multiple of 128 -> no boundary tiles
#define N_EDGES 1600000            // multiple of TEDGE -> no boundary tiles
#define N_CONF  2000
#define N_MOL   500
#define HD      128
#define FD      128
#define GD      50
#define LAYERS  6
#define TEDGE   256

typedef __bf16 bf16_t;
typedef __attribute__((ext_vector_type(16))) __bf16 v16bf;
typedef __attribute__((ext_vector_type(8)))  __bf16 v8bf;
typedef __attribute__((ext_vector_type(4)))  __bf16 v4bf;
typedef __attribute__((ext_vector_type(8)))  float  v8f;
typedef __attribute__((ext_vector_type(4)))  float  f32x4;

// float -> bf16 bits, round-to-nearest-even
__device__ inline unsigned short f2bfu(float f){
  unsigned u = __float_as_uint(f);
  return (unsigned short)((u + 0x7FFFu + ((u >> 16) & 1u)) >> 16);
}
__device__ inline bf16_t f2bf(float f){
  union { unsigned short u; bf16_t b; } cv;
  cv.u = f2bfu(f);
  return cv.b;
}
__device__ inline float bfu2f(unsigned short u){
  return __uint_as_float((unsigned)u << 16);
}

__device__ inline float sspf(float v){
  float sp = (v > 20.f) ? v : log1pf(__expf(v));
  return sp - 0.69314718056f;   // softplus(x) - log(2)
}

__device__ inline v8f wmma_bf16(v16bf a, v16bf b, v8f c){
  return __builtin_amdgcn_wmma_f32_16x16x32_bf16(false, a, false, b, (short)0, c, false, false);
}

// Coalesced staging of a K x 128 f32 weight matrix into LDS, pre-swizzled into
// this file's bf16 B-fragment order (contiguous k per lane half): entry ((nTile*KB + kb)*32 + lane)*16 + j
// holds B[kb*32 + 16*(lane>>4) + j][nTile*16 + (lane&15)].
__device__ inline void stage_wswz(const float* __restrict__ Wg, bf16_t* sw,
                                  int KB, int Krows, int tid){
  int total = KB * 32 * 128;
  for (int i = tid; i < total; i += 256){
    int k = i >> 7, n = i & 127;
    float w = (k < Krows) ? Wg[k * 128 + n] : 0.f;
    int kb   = k >> 5;
    int r    = k & 31;
    int lane = ((r >> 4) << 4) | (n & 15);
    int j    = r & 15;
    int nt   = n >> 4;
    sw[(((nt * KB + kb) * 32 + lane) << 4) + j] = f2bf(w);
  }
}

// A-fragment (16x32 bf16) from row-major LDS: lane l holds row M=mBase+(l&15),
// elems 0..15 = K[khalf..+15], khalf = 16*(l>>4).
__device__ inline v16bf load_afrag(const bf16_t* sa, int ld, int mBase, int kBase, int lane){
  int M = mBase + (lane & 15);
  int khalf = (lane >> 4) * 16;
  const bf16_t* p = sa + M * ld + kBase + khalf;
  v8bf lo = *(const v8bf*)(p);
  v8bf hi = *(const v8bf*)(p + 8);
  v16bf r;
#pragma unroll
  for (int i = 0; i < 8; ++i){ r[i] = lo[i]; r[i + 8] = hi[i]; }
  return r;
}

__device__ inline v16bf load_bfrag(const bf16_t* sw, int KB, int nTile, int kb, int lane){
  const bf16_t* p = sw + (((nTile * KB + kb) * 32 + lane) * 16);
  v8bf lo = *(const v8bf*)(p);
  v8bf hi = *(const v8bf*)(p + 8);
  v16bf r;
#pragma unroll
  for (int i = 0; i < 8; ++i){ r[i] = lo[i]; r[i + 8] = hi[i]; }
  return r;
}

// Stage a full 128-row f32 tile as bf16 row-major into LDS (float4 loads)
__device__ inline void stage_atile(const float* __restrict__ in, bf16_t* sa,
                                   int rBase, int tid){
  const float* src = in + (size_t)rBase * 128;
  for (int i = tid * 4; i < 128 * 128; i += 256 * 4){
    f32x4 v = *(const f32x4*)(src + i);
    v4bf o; o[0] = f2bf(v.x); o[1] = f2bf(v.y); o[2] = f2bf(v.z); o[3] = f2bf(v.w);
    *(v4bf*)(sa + i) = o;
  }
}

// ---------------- small utility kernels ----------------

__global__ void edge_geom_kernel(const float* __restrict__ pos, const int* __restrict__ eidx,
                                 float* __restrict__ dvec, float* __restrict__ cvec){
  int e = blockIdx.x * blockDim.x + threadIdx.x;
  if (e >= N_EDGES) return;
  int s = eidx[e], t = eidx[N_EDGES + e];
  float dx = pos[s*3+0] - pos[t*3+0];
  float dy = pos[s*3+1] - pos[t*3+1];
  float dz = pos[s*3+2] - pos[t*3+2];
  float d = sqrtf(dx*dx + dy*dy + dz*dz + 1e-12f);
  dvec[e] = d;
  cvec[e] = 0.5f * (cosf(d * 0.62831853071795864f) + 1.0f);  // pi/5
}

__global__ void gather_emb_kernel(const int* __restrict__ z, const float* __restrict__ emb,
                                  float* __restrict__ h){
  long long i = (long long)blockIdx.x * blockDim.x + threadIdx.x;
  if (i >= (long long)N_ATOMS * 128) return;
  int row = (int)(i >> 7), col = (int)(i & 127);
  h[i] = emb[(long long)z[row] * 128 + col];
}

__global__ void zero_kernel(f32x4* __restrict__ p, long long n4){
  long long i = (long long)blockIdx.x * blockDim.x + threadIdx.x;
  long long stride = (long long)gridDim.x * blockDim.x;
  f32x4 zv = {0.f, 0.f, 0.f, 0.f};
  for (; i < n4; i += stride) p[i] = zv;
}

__global__ void seg_sum_kernel(const float* __restrict__ in, const int* __restrict__ seg,
                               float* __restrict__ out, int nrows){
  long long i = (long long)blockIdx.x * blockDim.x + threadIdx.x;
  long long stride = (long long)gridDim.x * blockDim.x;
  long long total = (long long)nrows * 128;
  for (; i < total; i += stride){
    int row = (int)(i >> 7), col = (int)(i & 127);
    atomicAdd(&out[(long long)seg[row] * 128 + col], in[i]);
  }
}

__global__ __launch_bounds__(64) void head_kernel(
    const float* __restrict__ mol, const float* __restrict__ h1w, const float* __restrict__ h1b,
    const float* __restrict__ h2w, const float* __restrict__ h2b, float* __restrict__ pred){
  __shared__ float srow[128];
  __shared__ float t1[64];
  int m = blockIdx.x, tid = threadIdx.x;
  srow[tid]      = mol[m * 128 + tid];
  srow[tid + 64] = mol[m * 128 + tid + 64];
  __syncthreads();
  float acc = 0.f;
#pragma unroll 4
  for (int i = 0; i < 128; ++i) acc += srow[i] * h1w[i * 64 + tid];
  t1[tid] = sspf(acc + h1b[tid]);
  __syncthreads();
  if (tid == 0){
    float s = 0.f;
    for (int j = 0; j < 64; ++j) s += t1[j] * h2w[j];
    pred[m] = s + h2b[0];
  }
}

// ---------------- WMMA GEMM kernels (all tiles full; sizes compile-time) ----------------

// xbf = in @ Wg   (N_PAD x 128) @ (128 x 128), bf16 output (halves edge-gather traffic)
__global__ __launch_bounds__(256) void gemm_node_kernel(
    const float* __restrict__ in, const float* __restrict__ Wg,
    unsigned short* __restrict__ xbf){
  extern __shared__ __align__(32) unsigned char smem[];
  bf16_t* sW = (bf16_t*)smem;          // 16384
  bf16_t* sA = sW + 16384;             // 16384
  int tid = threadIdx.x, lane = tid & 31, wave = tid >> 5;
  int rBase = blockIdx.x * 128;
  stage_wswz(Wg, sW, 4, 128, tid);
  stage_atile(in, sA, rBase, tid);
  __syncthreads();

  v16bf bf[4];
#pragma unroll
  for (int kb = 0; kb < 4; ++kb) bf[kb] = load_bfrag(sW, 4, wave, kb, lane);

  int mloc0 = ((lane >> 4) << 3);
  int col = wave * 16 + (lane & 15);
  for (int m = 0; m < 8; ++m){
    v8f acc = {};
#pragma unroll
    for (int kb = 0; kb < 4; ++kb)
      acc = wmma_bf16(load_afrag(sA, 128, m * 16, kb * 32, lane), bf[kb], acc);
    unsigned short* o = xbf + (size_t)(rBase + m * 16 + mloc0) * 128 + col;
#pragma unroll
    for (int r = 0; r < 8; ++r) o[(size_t)r * 128] = f2bfu(acc[r]);
  }
}

// out = ssp(in@W1 + B1) @ W2 + B2 (+ resid)
template <bool RESID>
__global__ __launch_bounds__(256) void fused_mlp2_kernel(
    const float* __restrict__ in,
    const float* __restrict__ W1, const float* __restrict__ B1,
    const float* __restrict__ W2, const float* __restrict__ B2,
    const float* __restrict__ resid, float* __restrict__ out){
  extern __shared__ __align__(32) unsigned char smem[];
  bf16_t* sW1 = (bf16_t*)smem;         // 16384
  bf16_t* sW2 = sW1 + 16384;           // 16384
  bf16_t* sA  = sW2 + 16384;           // 16384
  bf16_t* sY  = sA  + 16384;           // 16384
  __shared__ float sB1[128], sB2[128];
  int tid = threadIdx.x, lane = tid & 31, wave = tid >> 5;
  int rBase = blockIdx.x * 128;
  stage_wswz(W1, sW1, 4, 128, tid);
  stage_wswz(W2, sW2, 4, 128, tid);
  if (tid < 128){ sB1[tid] = B1[tid]; sB2[tid] = B2[tid]; }
  stage_atile(in, sA, rBase, tid);
  __syncthreads();

  int mloc0 = ((lane >> 4) << 3);
  int col = wave * 16 + (lane & 15);
  {
    v16bf bf[4];
#pragma unroll
    for (int kb = 0; kb < 4; ++kb) bf[kb] = load_bfrag(sW1, 4, wave, kb, lane);
    float b1 = sB1[col];
    for (int m = 0; m < 8; ++m){
      v8f acc = {};
#pragma unroll
      for (int kb = 0; kb < 4; ++kb)
        acc = wmma_bf16(load_afrag(sA, 128, m * 16, kb * 32, lane), bf[kb], acc);
#pragma unroll
      for (int r = 0; r < 8; ++r)
        sY[(m * 16 + mloc0 + r) * 128 + col] = f2bf(sspf(acc[r] + b1));
    }
  }
  __syncthreads();
  {
    v16bf bf[4];
#pragma unroll
    for (int kb = 0; kb < 4; ++kb) bf[kb] = load_bfrag(sW2, 4, wave, kb, lane);
    float b2 = sB2[col];
    for (int m = 0; m < 8; ++m){
      v8f acc = {};
#pragma unroll
      for (int kb = 0; kb < 4; ++kb)
        acc = wmma_bf16(load_afrag(sY, 128, m * 16, kb * 32, lane), bf[kb], acc);
      size_t base = (size_t)(rBase + m * 16 + mloc0) * 128 + col;
      if (RESID){
#pragma unroll
        for (int r = 0; r < 8; ++r)
          out[base + (size_t)r * 128] = acc[r] + b2 + resid[base + (size_t)r * 128];
      } else {
#pragma unroll
        for (int r = 0; r < 8; ++r)
          out[base + (size_t)r * 128] = acc[r] + b2;
      }
    }
  }
}

// Fused edge pipeline (256 edges/block, E % 256 == 0):
// edge_attr -> ssp(attr@iw1+b1) -> @iw2+b2 -> *C -> *x[src](bf16) -> atomicAdd agg[dst]
__global__ __launch_bounds__(256) void edge_msg_kernel(
    const float* __restrict__ dvec, const float* __restrict__ cvec,
    const int* __restrict__ eidx, const unsigned short* __restrict__ xbf,
    const float* __restrict__ W1g, const float* __restrict__ B1g,
    const float* __restrict__ W2g, const float* __restrict__ B2g,
    float* __restrict__ agg){
  extern __shared__ __align__(32) unsigned char smem[];
  bf16_t* sW1 = (bf16_t*)smem;         // 8192
  bf16_t* sW2 = sW1 + 8192;            // 16384
  bf16_t* sA  = sW2 + 16384;           // 256*64  = 16384
  bf16_t* sY  = sA  + 16384;           // 256*128 = 32768
  __shared__ float sB1[128], sB2[128], sC[TEDGE], sD[TEDGE];
  __shared__ int sSrc[TEDGE], sDst[TEDGE];
  int tid = threadIdx.x, lane = tid & 31, wave = tid >> 5;
  int eBase = blockIdx.x * TEDGE;

  stage_wswz(W1g, sW1, 2, GD, tid);    // 50x128, K padded to 64 with zeros
  stage_wswz(W2g, sW2, 4, 128, tid);
  if (tid < 128){ sB1[tid] = B1g[tid]; sB2[tid] = B2g[tid]; }
  {
    int e = eBase + tid;
    sD[tid] = dvec[e]; sC[tid] = cvec[e];
    sSrc[tid] = eidx[e]; sDst[tid] = eidx[N_EDGES + e];
  }
  __syncthreads();

  // Gaussian smearing in-register -> bf16 LDS tile (TEDGE x 64, cols 50..63 = 0)
  const float step  = 5.0f / 49.0f;
  const float coeff = -0.5f / (step * step);
  for (int q = tid; q < TEDGE * 16; q += 256){
    int r = q >> 4, g0 = (q & 15) * 4;
    float dv = sD[r];
    v4bf o;
#pragma unroll
    for (int u = 0; u < 4; ++u){
      int g = g0 + u;
      float val = 0.f;
      if (g < GD){
        float t = dv - step * (float)g;
        val = __expf(coeff * t * t);
      }
      o[u] = f2bf(val);
    }
    *(v4bf*)(sA + r * 64 + g0) = o;
  }
  __syncthreads();

  int mloc0 = ((lane >> 4) << 3);
  int col = wave * 16 + (lane & 15);

  // GEMM1: (TEDGE x 64) @ (64 x 128) -> ssp -> sY (bf16)
  {
    v16bf bf[2];
#pragma unroll
    for (int kb = 0; kb < 2; ++kb) bf[kb] = load_bfrag(sW1, 2, wave, kb, lane);
    float b1 = sB1[col];
    for (int m = 0; m < TEDGE / 16; ++m){
      v8f acc = {};
#pragma unroll
      for (int kb = 0; kb < 2; ++kb)
        acc = wmma_bf16(load_afrag(sA, 64, m * 16, kb * 32, lane), bf[kb], acc);
#pragma unroll
      for (int r = 0; r < 8; ++r)
        sY[(m * 16 + mloc0 + r) * 128 + col] = f2bf(sspf(acc[r] + b1));
    }
  }
  __syncthreads();

  // GEMM2 + epilogue: *C, gather x[src] (bf16), atomic scatter to agg[dst]
  {
    v16bf bf[4];
#pragma unroll
    for (int kb = 0; kb < 4; ++kb) bf[kb] = load_bfrag(sW2, 4, wave, kb, lane);
    float b2 = sB2[col];
    for (int m = 0; m < TEDGE / 16; ++m){
      v8f acc = {};
#pragma unroll
      for (int kb = 0; kb < 4; ++kb)
        acc = wmma_bf16(load_afrag(sY, 128, m * 16, kb * 32, lane), bf[kb], acc);
      int el0 = m * 16 + mloc0;
#pragma unroll
      for (int r = 0; r < 8; ++r){
        int el = el0 + r;
        float w  = (acc[r] + b2) * sC[el];
        float xv = bfu2f(xbf[(size_t)sSrc[el] * 128 + col]);
        atomicAdd(&agg[(size_t)sDst[el] * 128 + col], w * xv);
      }
    }
  }
}

extern "C" void kernel_launch(void* const* d_in, const int* in_sizes, int n_in,
                              void* d_out, int out_size, void* d_ws, size_t ws_size,
                              hipStream_t stream){
  (void)in_sizes; (void)n_in; (void)out_size; (void)ws_size;
  const int*   z    = (const int*)  d_in[0];
  const float* pos  = (const float*)d_in[1];
  const int*   eidx = (const int*)  d_in[2];
  const int*   a2c  = (const int*)  d_in[3];
  const int*   c2m  = (const int*)  d_in[4];
  const float* emb  = (const float*)d_in[5];
  const float* iw1  = (const float*)d_in[6];
  const float* ib1  = (const float*)d_in[7];
  const float* iw2  = (const float*)d_in[8];
  const float* ib2  = (const float*)d_in[9];
  const float* il1  = (const float*)d_in[10];
  const float* il2w = (const float*)d_in[11];
  const float* il2b = (const float*)d_in[12];
  const float* ilw  = (const float*)d_in[13];
  const float* ilb  = (const float*)d_in[14];
  const float* l1w  = (const float*)d_in[15];
  const float* l1b  = (const float*)d_in[16];
  const float* l2w  = (const float*)d_in[17];
  const float* l2b  = (const float*)d_in[18];
  const float* h1w  = (const float*)d_in[19];
  const float* h1b  = (const float*)d_in[20];
  const float* h2w  = (const float*)d_in[21];
  const float* h2b  = (const float*)d_in[22];

  float* ws   = (float*)d_ws;
  float* dvec = ws;                                   // E
  float* cvec = dvec + N_EDGES;                       // E
  float* h    = cvec + N_EDGES;                       // N_PAD*128 f32
  float* agg  = h    + (size_t)N_PAD * 128;           // N_PAD*128 f32 (also final hfin)
  unsigned short* xbf = (unsigned short*)(agg + (size_t)N_PAD * 128); // N_PAD*128 bf16
  float* conf = (float*)(xbf + (size_t)N_PAD * 128);  // 2000*128
  float* mol  = conf + (size_t)N_CONF * 128;          // 500*128

  const size_t shEdge = (size_t)(8192 + 16384 + 16384 + 32768) * sizeof(bf16_t); // 144 KB
  const size_t shMlp2 = (size_t)(16384 * 4) * sizeof(bf16_t);                    // 128 KB
  const size_t shGemm = (size_t)(16384 * 2) * sizeof(bf16_t);                    //  64 KB
  const int nodeBlocks = N_PAD / 128;      // 391
  const int edgeBlocks = N_EDGES / TEDGE;  // 6250

  edge_geom_kernel<<<(N_EDGES + 255) / 256, 256, 0, stream>>>(pos, eidx, dvec, cvec);
  gather_emb_kernel<<<((long long)N_ATOMS * 128 + 255) / 256, 256, 0, stream>>>(z, emb, h);
  // zero h padding rows once (deterministic padding)
  zero_kernel<<<8, 256, 0, stream>>>((f32x4*)(h + (size_t)N_ATOMS * 128),
                                     (long long)(N_PAD - N_ATOMS) * 128 / 4);

  for (int k = 0; k < LAYERS; ++k){
    gemm_node_kernel<<<nodeBlocks, 256, shGemm, stream>>>(h, il1 + (size_t)k * HD * FD, xbf);
    zero_kernel<<<1024, 256, 0, stream>>>((f32x4*)agg, (long long)N_PAD * 128 / 4);
    edge_msg_kernel<<<edgeBlocks, 256, shEdge, stream>>>(
        dvec, cvec, eidx, xbf,
        iw1 + (size_t)k * GD * FD, ib1 + (size_t)k * FD,
        iw2 + (size_t)k * FD * FD, ib2 + (size_t)k * FD,
        agg);
    fused_mlp2_kernel<true><<<nodeBlocks, 256, shMlp2, stream>>>(
        agg, il2w + (size_t)k * FD * HD, il2b + (size_t)k * HD,
        ilw + (size_t)k * HD * HD, ilb + (size_t)k * HD,
        h, h);
  }

  // final projection: hfin = ssp(h@l1w + l1b)@l2w + l2b  (into agg buffer)
  fused_mlp2_kernel<false><<<nodeBlocks, 256, shMlp2, stream>>>(h, l1w, l1b, l2w, l2b, nullptr, agg);

  zero_kernel<<<64, 256, 0, stream>>>((f32x4*)conf, (long long)N_CONF * 128 / 4);
  zero_kernel<<<16, 256, 0, stream>>>((f32x4*)mol,  (long long)N_MOL  * 128 / 4);
  seg_sum_kernel<<<1024, 256, 0, stream>>>(agg, a2c, conf, N_ATOMS);
  seg_sum_kernel<<<64,   256, 0, stream>>>(conf, c2m, mol, N_CONF);
  head_kernel<<<N_MOL, 64, 0, stream>>>(mol, h1w, h1b, h2w, h2b, (float*)d_out);
}
